// CausalAttention_35553739276994
// MI455X (gfx1250) — matrix operand order repaired and checked by reading, not compiled, not run on hardware
//
#include <hip/hip_runtime.h>

// CausalAttention for MI455X (gfx1250, wave32).
// Pipeline: (0) one-shot W transpose to bf16 (L2-resident, 384 KB),
//           (1) QKV projection GEMM (bf16 WMMA, fp32 accum), LDS-free main loop,
//           (2) Flash attention with online softmax, bf16 WMMA, fp32 statistics.
// Roofline: ~15 GFLOP total, ~75 MB HBM traffic; Q/K/Vt (6 MB bf16) and WT fit
// in the 192 MB L2, so all operand tiles stream from global; WMMA/L2-bound.
//
// Round-4 fix: kernel-1 x rows have no cross-wave reuse -> drop the LDS
// staging + barriers; A-frags convert fp32->bf16 inline, B-frags load
// contiguously from the pre-transposed bf16 W.

#define B_   4
#define S_   4096
#define DIN  1024
#define H_   64

typedef __bf16 bf16;
typedef bf16  bf16x8  __attribute__((ext_vector_type(8)));
typedef bf16  bf16x16 __attribute__((ext_vector_type(16)));
typedef float f32x4   __attribute__((ext_vector_type(4)));
typedef float f32x8   __attribute__((ext_vector_type(8)));

static __device__ __forceinline__ f32x8 wmma_bf16(bf16x16 a, bf16x16 b, f32x8 c) {
    // V_WMMA_F32_16X16X32_BF16: D = A(16x32) * B(32x16) + C, fp32 accumulate.
    return __builtin_amdgcn_wmma_f32_16x16x32_bf16(false, a, false, b, (short)0, c, false, false);
}

// Reduce across the 16-lane group that shares one row-set in the C/D layout
// (lanes 0-15 hold rows 0-7, lanes 16-31 hold rows 8-15; lane%16 = column).
static __device__ __forceinline__ float rowmax16(float v) {
    v = fmaxf(v, __shfl_xor(v, 1, 16));
    v = fmaxf(v, __shfl_xor(v, 2, 16));
    v = fmaxf(v, __shfl_xor(v, 4, 16));
    v = fmaxf(v, __shfl_xor(v, 8, 16));
    return v;
}
static __device__ __forceinline__ float rowsum16(float v) {
    v += __shfl_xor(v, 1, 16);
    v += __shfl_xor(v, 2, 16);
    v += __shfl_xor(v, 4, 16);
    v += __shfl_xor(v, 8, 16);
    return v;
}

// Assemble a 16-element A/B fragment from two contiguous 8-element runs.
static __device__ __forceinline__ bf16x16 frag_from_runs(const bf16* lo_p, const bf16* hi_p) {
    const bf16x8 lo = *(const bf16x8*)lo_p;
    const bf16x8 hi = *(const bf16x8*)hi_p;
    bf16x16 f;
#pragma unroll
    for (int i = 0; i < 8; ++i) { f[i] = lo[i]; f[8 + i] = hi[i]; }
    return f;
}

// ---------------------------------------------------------------------------
// Kernel 0: one-shot W transpose + bf16 convert: W[1024][64] fp32 ->
// WT[mat][64][1024] bf16. grid = (DIN/64, 3), block 256. Loads are coalesced
// across lanes; stores are 2x16B contiguous per thread.
// ---------------------------------------------------------------------------
__global__ __launch_bounds__(256) void transpose_w_kernel(
    const float* __restrict__ Wq, const float* __restrict__ Wk,
    const float* __restrict__ Wv, bf16* __restrict__ WT)
{
    const int mat = blockIdx.y;
    const float* W = (mat == 0) ? Wq : (mat == 1) ? Wk : Wv;
    const int k0   = blockIdx.x * 64;
    const int c    = threadIdx.x & 63;
    const int kloc = (threadIdx.x >> 6) * 16;

    bf16x8 v0, v1;
#pragma unroll
    for (int i = 0; i < 8; ++i) v0[i] = (bf16)W[(size_t)(k0 + kloc + i) * H_ + c];
#pragma unroll
    for (int i = 0; i < 8; ++i) v1[i] = (bf16)W[(size_t)(k0 + kloc + 8 + i) * H_ + c];

    bf16* dst = WT + (size_t)mat * H_ * DIN + (size_t)c * DIN + k0 + kloc;
    *(bf16x8*)dst       = v0;
    *(bf16x8*)(dst + 8) = v1;
}

// ---------------------------------------------------------------------------
// Kernel 1: fused QKV projection. grid = (S/128, B, 3), block = 256 (8 waves).
// LDS-free, barrier-free main loop: each wave owns 16 disjoint x rows.
// A-frag (16-bit, 16x32): lane m = row; k = (lane/16)*16 + i,
//   built inline from fp32 x. B-frag: lane n = col; k = (lane/16)*16 + i ->
//   contiguous in the pre-transposed WT[col][k].
// ---------------------------------------------------------------------------
__global__ __launch_bounds__(256) void qkv_proj_kernel(
    const float* __restrict__ x, const bf16* __restrict__ WT,
    bf16* __restrict__ Qb, bf16* __restrict__ Kb, bf16* __restrict__ Vt)
{
    __shared__ bf16 obuf[128][72];  // epilogue coalescing slab (per-wave rows)

    const int tid  = threadIdx.x;
    const int wave = tid >> 5, lane = tid & 31;
    const int lrow = lane & 15, lhalf = lane >> 4;
    const int b    = blockIdx.y;
    const int mat  = blockIdx.z;           // 0:Q 1:K 2:V
    const int r0   = blockIdx.x * 128;

    const float* xrow = x + ((size_t)b * S_ + (r0 + wave * 16 + lrow)) * DIN;
    const bf16*  WTm  = WT + (size_t)mat * H_ * DIN;

    f32x8 acc[4] = {};                     // 16 rows x 64 cols per wave

    for (int kt = 0; kt < DIN / 32; ++kt) {
        const int k0 = kt * 32;

        // A fragment: two 8-float runs of this lane's x row, converted inline.
        bf16x16 a;
        {
            const float* p0 = xrow + k0 + lhalf * 16;        // k = base..base+7
            const float* p1 = xrow + k0 + lhalf * 16 + 8;    // k = base+8..base+15
            const f32x4 a0 = *(const f32x4*)(p0 + 0);
            const f32x4 a1 = *(const f32x4*)(p0 + 4);
            const f32x4 a2 = *(const f32x4*)(p1 + 0);
            const f32x4 a3 = *(const f32x4*)(p1 + 4);
#pragma unroll
            for (int j = 0; j < 4; ++j) {
                a[j]      = (bf16)a0[j];
                a[4 + j]  = (bf16)a1[j];
                a[8 + j]  = (bf16)a2[j];
                a[12 + j] = (bf16)a3[j];
            }
        }

        // B fragments: contiguous 32B per lane from L2-resident WT.
        bf16x16 bfr[4];
#pragma unroll
        for (int n = 0; n < 4; ++n) {
            const bf16* wp = WTm + (size_t)(n * 16 + lrow) * DIN + k0 + lhalf * 16;
            bfr[n] = frag_from_runs(wp, wp + 8);
        }
#pragma unroll
        for (int n = 0; n < 4; ++n) acc[n] = wmma_bf16(a, bfr[n], acc[n]);
    }

    // Epilogue: uniform branch on mat, compile-time strides in each arm.
    if (mat == 2) {
        // V transposed: one D tile's 8 accumulator slots are 8 consecutive seq
        // positions -> pack to bf16x8, 4 x 16B contiguous stores per lane.
        bf16* dstv = Vt + ((size_t)b * H_ + lrow) * S_ + (r0 + wave * 16 + lhalf * 8);
#pragma unroll
        for (int n = 0; n < 4; ++n) {
            bf16x8 pk;
#pragma unroll
            for (int r = 0; r < 8; ++r) pk[r] = (bf16)acc[n][r];
            *(bf16x8*)(dstv + (size_t)(n * 16) * S_) = pk;
        }
    } else {
        // Q/K row-major: coalesce via per-wave LDS slab (same-wave, in-order).
#pragma unroll
        for (int n = 0; n < 4; ++n)
#pragma unroll
            for (int r = 0; r < 8; ++r)
                obuf[wave * 16 + lhalf * 8 + r][n * 16 + lrow] = (bf16)acc[n][r];

        bf16* dstqk = (mat == 0) ? Qb : Kb;
        const int orow = wave * 16 + lrow;          // lane covers one row half
        const bf16* srow = &obuf[orow][lhalf * 32];
        bf16* drow = dstqk + ((size_t)b * S_ + r0 + orow) * H_ + lhalf * 32;
#pragma unroll
        for (int j = 0; j < 4; ++j)
            *(bf16x8*)(drow + j * 8) = *(const bf16x8*)(srow + j * 8);
    }
}

// ---------------------------------------------------------------------------
// Kernel 2: flash attention. grid = (S/128, B), block = 256 (8 waves),
// each wave owns 16 query rows and iterates causal 64-key blocks.
// ---------------------------------------------------------------------------
__global__ __launch_bounds__(256) void flash_attn_kernel(
    const bf16* __restrict__ Qb, const bf16* __restrict__ Kb,
    const bf16* __restrict__ Vt, float* __restrict__ Out)
{
    __shared__ bf16  pbuf[8][16][72];    // per-wave P transpose slab (D->A layout)
    __shared__ float obuf[8][16][68];    // per-wave output coalescing slab

    const int tid  = threadIdx.x;
    const int wave = tid >> 5, lane = tid & 31;
    const int lrow = lane & 15, lhalf = lane >> 4;
    const int b    = blockIdx.y;
    const int q0   = blockIdx.x * 128 + wave * 16;

    // Q fragments pinned in VGPRs for the whole kernel (head dim = 2 K-steps)
    bf16x16 aq[2];
    {
        const bf16* qrow = Qb + ((size_t)b * S_ + q0 + lrow) * H_;
#pragma unroll
        for (int ks = 0; ks < 2; ++ks)
            aq[ks] = frag_from_runs(qrow + ks * 32 + lhalf * 16,
                                    qrow + ks * 32 + lhalf * 16 + 8);
    }

    f32x8 acc[4] = {};
    float m_i[8], l_i[8];
#pragma unroll
    for (int r = 0; r < 8; ++r) { m_i[r] = -3.0e38f; l_i[r] = 0.0f; }

    const float scale = 0.125f;                 // 1/sqrt(64)
    const int   nkb   = (q0 + 15) / 64 + 1;     // causal: skip fully-masked blocks

    for (int kt = 0; kt < nkb; ++kt) {
        const int k0 = kt * 64;
        const bool needs_mask = (k0 + 63) > q0;  // wave-uniform: trailing block(s)

        if (kt + 1 < nkb)  // hint next K block toward L2/L0
            __builtin_prefetch(Kb + ((size_t)b * S_ + (k0 + 64 + lane)) * H_, 0, 1);

        // ---- S = Q * K^T: issue all 8 K-frag loads, then 8 WMMAs ----
        bf16x16 bk[4][2];
#pragma unroll
        for (int n = 0; n < 4; ++n) {
            const bf16* krow = Kb + ((size_t)b * S_ + (k0 + n * 16 + lrow)) * H_ + lhalf * 16;
#pragma unroll
            for (int ks = 0; ks < 2; ++ks)
                bk[n][ks] = *(const bf16x16*)(krow + ks * 32);
        }
        f32x8 sacc[4] = {};
#pragma unroll
        for (int n = 0; n < 4; ++n)
#pragma unroll
            for (int ks = 0; ks < 2; ++ks)
                sacc[n] = wmma_bf16(aq[ks], bk[n][ks], sacc[n]);

        // ---- scale (+ causal mask only on trailing block) + softmax stats ----
        float mnew[8];
#pragma unroll
        for (int r = 0; r < 8; ++r) {
            float tmax = -3.0e38f;
#pragma unroll
            for (int n = 0; n < 4; ++n) {
                float s = sacc[n][r] * scale;
                if (needs_mask) {
                    const int grow = q0 + lhalf * 8 + r;
                    const int gcol = k0 + n * 16 + lrow;
                    s = (gcol <= grow) ? s : -3.0e38f;
                }
                sacc[n][r] = s;
                tmax = fmaxf(tmax, s);
            }
            tmax    = rowmax16(tmax);
            mnew[r] = fmaxf(m_i[r], tmax);
        }

        float alpha[8];
#pragma unroll
        for (int r = 0; r < 8; ++r) {
            alpha[r] = __expf(m_i[r] - mnew[r]);
            m_i[r]   = mnew[r];
            float rs = 0.0f;
#pragma unroll
            for (int n = 0; n < 4; ++n) {
                const float p = __expf(sacc[n][r] - mnew[r]);
                rs += p;
                // D layout -> LDS (row-major P); same-wave LDS is in-order
                pbuf[wave][lhalf * 8 + r][n * 16 + lrow] = (bf16)p;
            }
            rs     = rowsum16(rs);
            l_i[r] = l_i[r] * alpha[r] + rs;
        }

        // rescale O accumulators by alpha (per row = per accumulator slot)
#pragma unroll
        for (int n = 0; n < 4; ++n)
#pragma unroll
            for (int r = 0; r < 8; ++r) acc[n][r] *= alpha[r];

        // reload P in A layout from LDS (contiguous runs per lane)
        bf16x16 ap[2];
        {
            const bf16* prow = &pbuf[wave][lrow][0];
#pragma unroll
            for (int ks = 0; ks < 2; ++ks)
                ap[ks] = frag_from_runs(prow + ks * 32 + lhalf * 16,
                                        prow + ks * 32 + lhalf * 16 + 8);
        }

        // ---- O += P * V: issue all 8 V-frag loads, then 8 WMMAs ----
        bf16x16 bv[4][2];
#pragma unroll
        for (int n = 0; n < 4; ++n) {
            const bf16* vrow = Vt + ((size_t)b * H_ + (n * 16 + lrow)) * S_ + k0 + lhalf * 16;
#pragma unroll
            for (int ks = 0; ks < 2; ++ks)
                bv[n][ks] = *(const bf16x16*)(vrow + ks * 32);
        }
#pragma unroll
        for (int n = 0; n < 4; ++n)
#pragma unroll
            for (int ks = 0; ks < 2; ++ks)
                acc[n] = wmma_bf16(ap[ks], bv[n][ks], acc[n]);
    }

    // ---- normalize; coalesce output through per-wave LDS slab ----
#pragma unroll
    for (int r = 0; r < 8; ++r) {
        const float inv = 1.0f / l_i[r];
#pragma unroll
        for (int n = 0; n < 4; ++n)
            obuf[wave][lhalf * 8 + r][n * 16 + lrow] = acc[n][r] * inv;
    }
    {   // lane writes 128 contiguous bytes of one query row
        const float* srow = &obuf[wave][lrow][lhalf * 32];
        float* drow = Out + ((size_t)b * S_ + q0 + lrow) * H_ + lhalf * 32;
#pragma unroll
        for (int j = 0; j < 8; ++j)
            *(f32x4*)(drow + j * 4) = *(const f32x4*)(srow + j * 4);
    }
}

extern "C" void kernel_launch(void* const* d_in, const int* in_sizes, int n_in,
                              void* d_out, int out_size, void* d_ws, size_t ws_size,
                              hipStream_t stream) {
    const float* x  = (const float*)d_in[0];
    const float* Wq = (const float*)d_in[1];
    const float* Wk = (const float*)d_in[2];
    const float* Wv = (const float*)d_in[3];
    float* out = (float*)d_out;

    // Workspace: bf16 Q [B,S,H], K [B,S,H], V^T [B,H,S], WT [3][H][DIN]
    bf16* Qb = (bf16*)d_ws;
    bf16* Kb = Qb + (size_t)B_ * S_ * H_;
    bf16* Vt = Kb + (size_t)B_ * S_ * H_;
    bf16* WT = Vt + (size_t)B_ * S_ * H_;

    transpose_w_kernel<<<dim3(DIN / 64, 3), 256, 0, stream>>>(Wq, Wk, Wv, WT);
    qkv_proj_kernel<<<dim3(S_ / 128, B_, 3), 256, 0, stream>>>(x, WT, Qb, Kb, Vt);
    flash_attn_kernel<<<dim3(S_ / 128, B_), 256, 0, stream>>>(Qb, Kb, Vt, out);
}
